// VRSLDS_60155311948273
// MI455X (gfx1250) — matrix operand order repaired and checked by reading, not compiled, not run on hardware
//
#include <hip/hip_runtime.h>
#include <hip/hip_bf16.h>

typedef __attribute__((ext_vector_type(16))) _Float16 v16h;
typedef __attribute__((ext_vector_type(8)))  _Float16 v8h;
typedef __attribute__((ext_vector_type(8)))  float    v8f;

#define B_   64
#define T_   1024
#define K_   8
#define D_   32
#define O_   64
#define NTILES (B_ * (T_ / 16))   // 4096 16-token tiles

// LDS layout (halfword offsets). Each B-operand = 32 lanes * 16 halfs = 512 halfs = 1KB.
#define CW_OFF 0        // 32 operands (k*4+n)
#define AW_OFF 16384    // 16 operands (k*2+n)
#define RW_OFF 24576    //  8 operands (k)
#define CB_OFF 28672    //  4 operands (n)
#define AB_OFF 30720    //  2 operands (n)
#define RB_OFF 31744    //  1 operand
#define WLDS_HALFS 32256  // 63 KB

__device__ __forceinline__ v16h lds_op(const _Float16* wlds, int off, int lane) {
  return *(const v16h*)(wlds + off + lane * 16);  // 32B per lane, 32B aligned
}

__device__ __forceinline__ v16h hscale(v16h a, float s) {
  _Float16 h = (_Float16)s;
  v16h r;
#pragma unroll
  for (int i = 0; i < 16; ++i) r[i] = a[i] * h;   // -> v_pk_mul_f16
  return r;
}

__device__ __forceinline__ v8h cvt8(const float* __restrict__ src) {
  float4 f0 = *(const float4*)src;
  float4 f1 = *(const float4*)(src + 4);
  v8h h;
  h[0] = (_Float16)f0.x; h[1] = (_Float16)f0.y; h[2] = (_Float16)f0.z; h[3] = (_Float16)f0.w;
  h[4] = (_Float16)f1.x; h[5] = (_Float16)f1.y; h[6] = (_Float16)f1.z; h[7] = (_Float16)f1.w;
  return h;
}

#define WMMA(Aop, Boff, C) \
  __builtin_amdgcn_wmma_f32_16x16x32_f16(false, (Aop), false, \
      lds_op(wlds, (Boff), lane), (short)0, (C), false, false)

__global__ __launch_bounds__(256) void vrslds_wmma_kernel(
    const float* __restrict__ x,  const float* __restrict__ z,
    const float* __restrict__ Cw, const float* __restrict__ Cb,
    const float* __restrict__ Aw, const float* __restrict__ Ab,
    const float* __restrict__ Rw, const float* __restrict__ Rb,
    float* __restrict__ out_y, float* __restrict__ out_z, float* __restrict__ out_x)
{
  __shared__ __align__(32) _Float16 wlds[WLDS_HALFS];
  const int tid  = threadIdx.x;
  const int lane = tid & 31;
  const int wave = tid >> 5;

  // One tile per wave, straight-line (no loop -> no LICM of LDS operands).
  const int tile = (int)blockIdx.x * 8 + wave;     // grid = 512 blocks -> tile < 4096
  const int b   = tile >> 6;
  const int t0  = (tile & 63) << 4;
  const int row = lane & 15;   // token row within tile (A layout: M = lane&15)
  const int hs  = lane >> 4;   // half-select (A layout K-group / C row-group)
  const int t   = t0 + row;

  const float* xr = x + ((size_t)(b * T_ + t)) * D_;
  const float* zr = z + ((size_t)(b * T_ + t)) * K_;
  __builtin_prefetch(xr, 0, 0);                    // global_prefetch_b8, in flight
  __builtin_prefetch(zr, 0, 0);                    // during weight staging

  // ---- Stage weights into LDS (f32 -> f16), pre-swizzled to WMMA B layout ----
  // Operand element [l][j]: n-col = l&15, k-row = (l>>4)*16 + j. 8 halfs per step.
  for (int c = tid; c < 2048; c += 256) {          // Cw (K,O,D): op = k*4+n
    int op = c >> 6, rem = c & 63, l = rem >> 1, h = rem & 1;
    int k = op >> 2, n = op & 3;
    int o  = n * 16 + (l & 15);
    int d0 = ((l >> 4) << 4) + h * 8;
    *(v8h*)(wlds + CW_OFF + (op * 32 + l) * 16 + h * 8) =
        cvt8(Cw + (k * O_ + o) * D_ + d0);
  }
  for (int c = tid; c < 1024; c += 256) {          // Aw (K,D,D): op = k*2+n
    int op = c >> 6, rem = c & 63, l = rem >> 1, h = rem & 1;
    int k = op >> 1, n = op & 1;
    int e  = n * 16 + (l & 15);
    int d0 = ((l >> 4) << 4) + h * 8;
    *(v8h*)(wlds + AW_OFF + (op * 32 + l) * 16 + h * 8) =
        cvt8(Aw + (k * D_ + e) * D_ + d0);
  }
  for (int c = tid; c < 512; c += 256) {           // Rw (K,K-1,D): op = k, pad r>=7
    int k = c >> 6, rem = c & 63, l = rem >> 1, h = rem & 1;
    int r  = l & 15;
    int d0 = ((l >> 4) << 4) + h * 8;
    v8h hv = {};
    if (r < K_ - 1) hv = cvt8(Rw + (k * (K_ - 1) + r) * D_ + d0);
    *(v8h*)(wlds + RW_OFF + (k * 32 + l) * 16 + h * 8) = hv;
  }
  for (int idx = tid; idx < 4 * 512; idx += 256) { // Cb (K,O): k-dim = regime
    int n = idx >> 9, rem = idx & 511, l = rem >> 4, j = rem & 15;
    int o = n * 16 + (l & 15), kr = ((l >> 4) << 4) + j;
    wlds[CB_OFF + idx] = (kr < K_) ? (_Float16)Cb[kr * O_ + o] : (_Float16)0.f;
  }
  for (int idx = tid; idx < 2 * 512; idx += 256) { // Ab (K,D)
    int n = idx >> 9, rem = idx & 511, l = rem >> 4, j = rem & 15;
    int e = n * 16 + (l & 15), kr = ((l >> 4) << 4) + j;
    wlds[AB_OFF + idx] = (kr < K_) ? (_Float16)Ab[kr * D_ + e] : (_Float16)0.f;
  }
  for (int idx = tid; idx < 512; idx += 256) {     // Rb (K,K-1)
    int l = (idx >> 4) & 31, j = idx & 15;
    int r = l & 15, kr = ((l >> 4) << 4) + j;
    wlds[RB_OFF + idx] = (kr < K_ && r < K_ - 1) ? (_Float16)Rb[kr * (K_ - 1) + r]
                                                 : (_Float16)0.f;
  }
  __syncthreads();

  // ---- A-tile: 16 tokens x 32 dims, f32 -> f16 in the 16-bit A-matrix layout ----
  float xa[8], xb[8];
  *(float4*)(&xa[0]) = *(const float4*)(xr + hs * 16);
  *(float4*)(&xa[4]) = *(const float4*)(xr + hs * 16 + 4);
  *(float4*)(&xb[0]) = *(const float4*)(xr + hs * 16 + 8);
  *(float4*)(&xb[4]) = *(const float4*)(xr + hs * 16 + 12);
  v16h ax;
#pragma unroll
  for (int i = 0; i < 8; ++i) { ax[i] = (_Float16)xa[i]; ax[8 + i] = (_Float16)xb[i]; }

  // Mixture weights: z[t] (y & z_lin paths), z[t+1] (x_next path)
  float zc[8], zn[8];
  *(float4*)(&zc[0]) = *(const float4*)(zr);
  *(float4*)(&zc[4]) = *(const float4*)(zr + 4);
  const int tn = (t + 1 < T_) ? (t + 1) : (T_ - 1);
  const float* zrn = z + ((size_t)(b * T_ + tn)) * K_;
  *(float4*)(&zn[0]) = *(const float4*)(zrn);
  *(float4*)(&zn[4]) = *(const float4*)(zrn + 4);

  // Bias A-operands: 16 tokens x 8 regimes, zero-padded to K=32.
  const _Float16 bm = (lane < 16) ? (_Float16)1.f : (_Float16)0.f;
  v16h abc = {}, abn = {};
#pragma unroll
  for (int j = 0; j < 8; ++j) {
    abc[j] = (_Float16)zc[j] * bm;
    abn[j] = (_Float16)zn[j] * bm;
  }

  const int col = lane & 15;

  // ======== Phase 1: y_pred (4 N-tiles) + z_lin, regimes folded into A ========
  {
    v8f y0 = {}, y1 = {}, y2 = {}, y3 = {}, zl = {};
#pragma unroll
    for (int k = 0; k < 8; ++k) {
      v16h ac = hscale(ax, zc[k]);               // rows scaled by z[t,k]
      y0 = WMMA(ac, CW_OFF + (k * 4 + 0) * 512, y0);
      y1 = WMMA(ac, CW_OFF + (k * 4 + 1) * 512, y1);
      y2 = WMMA(ac, CW_OFF + (k * 4 + 2) * 512, y2);
      y3 = WMMA(ac, CW_OFF + (k * 4 + 3) * 512, y3);
      zl = WMMA(ac, RW_OFF + k * 512, zl);
    }
    y0 = WMMA(abc, CB_OFF + 0 * 512, y0);
    y1 = WMMA(abc, CB_OFF + 1 * 512, y1);
    y2 = WMMA(abc, CB_OFF + 2 * 512, y2);
    y3 = WMMA(abc, CB_OFF + 3 * 512, y3);
    zl = WMMA(abc, RB_OFF, zl);

    // Stores (C/D layout: col = lane&15, rows m = hs*8 + j)
#pragma unroll
    for (int n = 0; n < 4; ++n) {
      const v8f acc = n == 0 ? y0 : n == 1 ? y1 : n == 2 ? y2 : y3;
#pragma unroll
      for (int j = 0; j < 8; ++j) {
        int m = hs * 8 + j;
        out_y[((size_t)(b * T_ + t0 + m)) * O_ + n * 16 + col] = acc[j];
      }
    }
    if (col < K_ - 1) {
#pragma unroll
      for (int j = 0; j < 8; ++j) {
        int tt = t0 + hs * 8 + j;
        if (tt < T_ - 1)
          out_z[((size_t)(b * (T_ - 1) + tt)) * K_ + col] = zl[j];  // raw z_lin
      }
    }
  }

  // ======== Phase 2: x_next (2 N-tiles), scaled by z[t+1,k] ========
  {
    v8f xn0 = {}, xn1 = {};
#pragma unroll
    for (int k = 0; k < 8; ++k) {
      v16h an = hscale(ax, zn[k]);
      xn0 = WMMA(an, AW_OFF + (k * 2 + 0) * 512, xn0);
      xn1 = WMMA(an, AW_OFF + (k * 2 + 1) * 512, xn1);
    }
    xn0 = WMMA(abn, AB_OFF + 0 * 512, xn0);
    xn1 = WMMA(abn, AB_OFF + 1 * 512, xn1);

#pragma unroll
    for (int n = 0; n < 2; ++n) {
      const v8f acc = n == 0 ? xn0 : xn1;
#pragma unroll
      for (int j = 0; j < 8; ++j) {
        int tt = t0 + hs * 8 + j;
        if (tt < T_ - 1)
          out_x[((size_t)(b * (T_ - 1) + tt)) * D_ + n * 16 + col] = acc[j];
      }
    }
  }
}

// Stick-breaking epilogue: in-place over the z_next slab (reads 7 logits, writes 8 probs).
__global__ __launch_bounds__(256) void vrslds_sb_kernel(float* __restrict__ zbuf, int ntok) {
  int i = (int)blockIdx.x * 256 + (int)threadIdx.x;
  if (i >= ntok) return;
  float* p = zbuf + (size_t)i * K_;
  float s[K_ - 1];
#pragma unroll
  for (int r = 0; r < K_ - 1; ++r) s[r] = 1.0f / (1.0f + __expf(-p[r]));
  float cum = 1.0f;
#pragma unroll
  for (int r = 0; r < K_ - 1; ++r) { p[r] = s[r] * cum; cum *= (1.0f - s[r]); }
  p[K_ - 1] = cum;
}

extern "C" void kernel_launch(void* const* d_in, const int* in_sizes, int n_in,
                              void* d_out, int out_size, void* d_ws, size_t ws_size,
                              hipStream_t stream) {
  (void)in_sizes; (void)n_in; (void)out_size; (void)d_ws; (void)ws_size;
  const float* x  = (const float*)d_in[0];
  const float* z  = (const float*)d_in[1];
  const float* Cw = (const float*)d_in[2];
  const float* Cb = (const float*)d_in[3];
  const float* Aw = (const float*)d_in[4];
  const float* Ab = (const float*)d_in[5];
  const float* Rw = (const float*)d_in[6];
  const float* Rb = (const float*)d_in[7];

  float* out_y = (float*)d_out;
  float* out_z = out_y + (size_t)B_ * T_ * O_;
  float* out_x = out_z + (size_t)B_ * (T_ - 1) * K_;

  // 512 blocks * 8 waves = 4096 waves = exactly one 16-token tile per wave.
  vrslds_wmma_kernel<<<NTILES / 8, 256, 0, stream>>>(x, z, Cw, Cb, Aw, Ab, Rw, Rb,
                                                     out_y, out_z, out_x);
  const int ntok = B_ * (T_ - 1);
  vrslds_sb_kernel<<<(ntok + 255) / 256, 256, 0, stream>>>(out_z, ntok);
}
